// GraphAttentionEmbedding_41583873360279
// MI455X (gfx1250) — matrix operand order repaired and checked by reading, not compiled, not run on hardware
//
#include <hip/hip_runtime.h>

typedef __attribute__((ext_vector_type(16))) _Float16 v16h;
typedef __attribute__((ext_vector_type(8)))  _Float16 v8h;
typedef __attribute__((ext_vector_type(8)))  float    v8f;

#define DIN  128
#define MSGD 128
#define HC   128   // H*C = 2*64

// ---- order-preserving float<->uint encoding for atomicMax-based segment max
__device__ __forceinline__ unsigned enc_f32(float f) {
  unsigned u = __float_as_uint(f);
  return (u & 0x80000000u) ? ~u : (u | 0x80000000u);
}
__device__ __forceinline__ float dec_f32(unsigned e) {
  unsigned u = (e & 0x80000000u) ? (e & 0x7FFFFFFFu) : ~e;
  return __uint_as_float(u);
}
#define ENC_NEG_INF 0x007FFFFFu   // enc(-inf)

__global__ void k_init(unsigned* __restrict__ menc, float* __restrict__ z, int n2) {
  int t = blockIdx.x * blockDim.x + threadIdx.x;
  if (t < n2) { menc[t] = ENC_NEG_INF; z[t] = 0.0f; }
}

// ------------------------------------------------------------------
// Prologue: convert the 5 weight matrices (f32 [k][col], 128x128 each) into
// f16 *transposed* [col][k] so a WMMA B-fragment is one contiguous 32B chunk.
// Order: Wq, Wk, Wv, Ws, We.
// ------------------------------------------------------------------
__global__ void k_cvt_w(const float* __restrict__ Wq, const float* __restrict__ Wk,
                        const float* __restrict__ Wv, const float* __restrict__ Ws,
                        const float* __restrict__ We, _Float16* __restrict__ wt)
{
  int t = blockIdx.x * blockDim.x + threadIdx.x;   // 5*128*128
  if (t >= 5 * DIN * HC) return;
  int m   = t >> 14;
  int idx = t & 16383;
  int k = idx >> 7;     // row of W
  int c = idx & 127;    // col of W
  const float* W = (m == 0) ? Wq : (m == 1) ? Wk : (m == 2) ? Wv : (m == 3) ? Ws : We;
  wt[((size_t)m * DIN * HC) + (size_t)c * DIN + k] = (_Float16)W[(size_t)k * HC + c];
}

// A fragment: two contiguous 8-half chunks of the LDS row, concatenated.
__device__ __forceinline__ v16h load_afrag(const _Float16* __restrict__ row,
                                           int k0, int kA) {
  v8h lo = *(const v8h*)(row + k0 + kA);
  v8h hi = *(const v8h*)(row + k0 + kA + 8);
  return __builtin_shufflevector(lo, hi,
      0, 1, 2, 3, 4, 5, 6, 7, 8, 9, 10, 11, 12, 13, 14, 15);
}

// ------------------------------------------------------------------
// Shared helper: 64x128 (f16 LDS tile) @ 128x128 (f16 transposed weights)
// -> 64x128 f32 LDS tile. One 16-col tile per wave, 4 row-tiles register-
// blocked so each B fragment is reused 4x. B frag = single 32B vector load.
// ------------------------------------------------------------------
__device__ __forceinline__ void gemm_tile_64x128(
    const _Float16 (*__restrict__ At)[MSGD + 8],
    const _Float16* __restrict__ Wt,          // [128 cols][128 k], f16
    float (*__restrict__ Dt)[HC + 4],
    int wave, int lane)
{
  const int mrow = lane & 15;
  const int kA   = (lane < 16) ? 0 : 16;    // 16-bit A-frag K offset
  const int kB   = (lane < 16) ? 0 : 16;    // B-frag K offset
  const int ncol = (wave << 4) + mrow;
  const _Float16* __restrict__ wp = Wt + (size_t)ncol * DIN;

  v8f acc[4] = {};
  #pragma unroll
  for (int k0 = 0; k0 < MSGD; k0 += 32) {
    v16h b = *(const v16h*)(wp + k0 + kB);   // 32B aligned: 2x global_load_b128
    #pragma unroll
    for (int rt = 0; rt < 4; ++rt) {
      v16h a = load_afrag(&At[rt * 16 + mrow][0], k0, kA);
      acc[rt] = __builtin_amdgcn_wmma_f32_16x16x32_f16(
          false, a, false, b, (short)0, acc[rt], false, false);
    }
  }
  #pragma unroll
  for (int rt = 0; rt < 4; ++rt) {
    #pragma unroll
    for (int i = 0; i < 8; ++i) {
      int r = rt * 16 + ((lane < 16) ? i : i + 8);
      Dt[r][ncol] = acc[rt][i];
    }
  }
}

// ------------------------------------------------------------------
// Kernel 1: fused q/k/v/skip node linears. 64 rows of x per block,
// 8 waves sweep the 4*128 = 512 output columns (32 16-col tiles).
// ------------------------------------------------------------------
__global__ __launch_bounds__(256) void k_node_linear(
    const float* __restrict__ x,
    const _Float16* __restrict__ wt,          // 4 transposed f16 matrices (+We)
    const float* __restrict__ bq, const float* __restrict__ bk,
    const float* __restrict__ bv, const float* __restrict__ bs,
    float* __restrict__ q, float* __restrict__ k, float* __restrict__ v,
    float* __restrict__ outp, int n)
{
  __shared__ _Float16 xt[64][DIN + 8];
  const int row0 = blockIdx.x * 64;
  const int tid  = threadIdx.x;

  for (int i = tid; i < 64 * DIN; i += 256) {
    int r = i >> 7, c = i & 127;
    int gr = row0 + r;
    xt[r][c] = (_Float16)((gr < n) ? x[(size_t)gr * DIN + c] : 0.0f);
  }
  __syncthreads();

  const int wave = tid >> 5, lane = tid & 31;
  const int mrow = lane & 15;
  const int kA   = (lane < 16) ? 0 : 16;
  const int kB   = (lane < 16) ? 0 : 16;

  for (int cti = 0; cti < 4; ++cti) {
    int ct  = wave + (cti << 3);        // 0..31
    int sel = ct >> 3;                  // which weight matrix
    int n0  = (ct & 7) << 4;            // column tile inside that matrix
    const float* B; float* O;
    if      (sel == 0) { B = bq; O = q; }
    else if (sel == 1) { B = bk; O = k; }
    else if (sel == 2) { B = bv; O = v; }
    else               { B = bs; O = outp; }
    const int ncol = n0 + mrow;
    const _Float16* __restrict__ wp =
        wt + ((size_t)sel * DIN * HC) + (size_t)ncol * DIN;

    v8f acc[4] = {};
    #pragma unroll
    for (int k0 = 0; k0 < DIN; k0 += 32) {
      v16h b = *(const v16h*)(wp + k0 + kB);
      #pragma unroll
      for (int rt = 0; rt < 4; ++rt) {
        v16h a = load_afrag(&xt[rt * 16 + mrow][0], k0, kA);
        acc[rt] = __builtin_amdgcn_wmma_f32_16x16x32_f16(
            false, a, false, b, (short)0, acc[rt], false, false);
      }
    }
    float bias = B[ncol];
    #pragma unroll
    for (int rt = 0; rt < 4; ++rt) {
      #pragma unroll
      for (int i = 0; i < 8; ++i) {
        int r = row0 + rt * 16 + ((lane < 16) ? i : i + 8);
        if (r < n) O[(size_t)r * HC + ncol] = acc[rt][i] + bias;
      }
    }
  }
}

// ------------------------------------------------------------------
// Kernel 2: edge pass A. ea = msg@We (WMMA, into LDS), attention logits,
// atomicMax segment max per (dst, head).
// ------------------------------------------------------------------
__global__ __launch_bounds__(256) void k_edge_logits(
    const float* __restrict__ msg, const _Float16* __restrict__ wte,
    const int* __restrict__ ei,
    const float* __restrict__ q, const float* __restrict__ k,
    float* __restrict__ logit, unsigned* __restrict__ menc, int E)
{
  __shared__ _Float16 mt[64][MSGD + 8];
  __shared__ float    ea[64][HC + 4];
  __shared__ int      ssrc[64], sdst[64];
  const int e0  = blockIdx.x * 64;
  const int tid = threadIdx.x;

  for (int i = tid; i < 64 * MSGD; i += 256) {
    int r = i >> 7, c = i & 127;
    int ge = e0 + r;
    mt[r][c] = (_Float16)((ge < E) ? msg[(size_t)ge * MSGD + c] : 0.0f);
  }
  if (tid < 64) {
    int ge = e0 + tid;
    ssrc[tid] = (ge < E) ? ei[ge] : 0;
    sdst[tid] = (ge < E) ? ei[(size_t)E + ge] : 0;
  }
  __syncthreads();

  const int wave = tid >> 5, lane = tid & 31;
  gemm_tile_64x128(mt, wte, ea, wave, lane);
  __syncthreads();

  // 64 edges * 2 heads = 128 (edge,head) pairs; 8 lanes each -> 4 sweeps.
  #pragma unroll
  for (int it = 0; it < 4; ++it) {
    int slot = it * 256 + tid;
    int sub  = slot & 7;
    int pair = slot >> 3;
    int le = pair >> 1, h = pair & 1;
    int src = ssrc[le], dst = sdst[le];
    int cb = h * 64 + sub * 8;
    float partial = 0.0f;
    #pragma unroll
    for (int j = 0; j < 8; ++j) {
      int ch = cb + j;
      partial += q[(size_t)dst * HC + ch] * (k[(size_t)src * HC + ch] + ea[le][ch]);
    }
    partial += __shfl_down(partial, 4, 8);
    partial += __shfl_down(partial, 2, 8);
    partial += __shfl_down(partial, 1, 8);
    if (sub == 0 && (e0 + le) < E) {
      float lg = partial * 0.125f;   // 1/sqrt(C), C=64
      logit[(size_t)(e0 + le) * 2 + h] = lg;
      atomicMax(&menc[(size_t)dst * 2 + h], enc_f32(lg));
    }
  }
}

// ------------------------------------------------------------------
// Kernel 3: p = exp(logit - m); z[dst,h] += p  (in-place over logit buffer)
// ------------------------------------------------------------------
__global__ void k_softmax_z(const int* __restrict__ ei,
                            const unsigned* __restrict__ menc,
                            float* __restrict__ logit,
                            float* __restrict__ z, int E)
{
  int t = blockIdx.x * blockDim.x + threadIdx.x;
  if (t >= 2 * E) return;
  int ge = t >> 1, h = t & 1;
  int dst = ei[(size_t)E + ge];
  float m = dec_f32(menc[(size_t)dst * 2 + h]);
  float p = __expf(logit[t] - m);
  logit[t] = p;
  unsafeAtomicAdd(&z[(size_t)dst * 2 + h], p);
}

// ------------------------------------------------------------------
// Kernel 4: edge pass B. Recompute ea (WMMA), scatter alpha * (v[src]+ea)
// into out[dst] with hardware f32 atomics.
// ------------------------------------------------------------------
__global__ __launch_bounds__(256) void k_edge_accum(
    const float* __restrict__ msg, const _Float16* __restrict__ wte,
    const int* __restrict__ ei, const float* __restrict__ v,
    const float* __restrict__ p, const float* __restrict__ z,
    float* __restrict__ outp, int E)
{
  __shared__ _Float16 mt[64][MSGD + 8];
  __shared__ float    ea[64][HC + 4];
  __shared__ int      ssrc[64], sdst[64];
  const int e0  = blockIdx.x * 64;
  const int tid = threadIdx.x;

  for (int i = tid; i < 64 * MSGD; i += 256) {
    int r = i >> 7, c = i & 127;
    int ge = e0 + r;
    mt[r][c] = (_Float16)((ge < E) ? msg[(size_t)ge * MSGD + c] : 0.0f);
  }
  if (tid < 64) {
    int ge = e0 + tid;
    ssrc[tid] = (ge < E) ? ei[ge] : 0;
    sdst[tid] = (ge < E) ? ei[(size_t)E + ge] : 0;
  }
  __syncthreads();

  const int wave = tid >> 5, lane = tid & 31;
  gemm_tile_64x128(mt, wte, ea, wave, lane);
  __syncthreads();

  // 64 edges * 16 channel-segments (8 ch each) = 1024 slots -> 4 sweeps.
  #pragma unroll
  for (int it = 0; it < 4; ++it) {
    int slot = it * 256 + tid;
    int le  = slot >> 4;
    int seg = slot & 15;
    int ge = e0 + le;
    if (ge >= E) continue;
    int src = ssrc[le], dst = sdst[le];
    int h = seg >> 3;     // channels [seg*8, seg*8+7] all belong to one head
    float w = p[(size_t)ge * 2 + h] / z[(size_t)dst * 2 + h];
    #pragma unroll
    for (int j = 0; j < 8; ++j) {
      int ch = seg * 8 + j;
      float vj = v[(size_t)src * HC + ch] + ea[le][ch];
      unsafeAtomicAdd(&outp[(size_t)dst * HC + ch], w * vj);
    }
  }
}

// ------------------------------------------------------------------
extern "C" void kernel_launch(void* const* d_in, const int* in_sizes, int n_in,
                              void* d_out, int out_size, void* d_ws, size_t ws_size,
                              hipStream_t stream) {
  const float* x   = (const float*)d_in[0];
  const int*   ei  = (const int*)d_in[1];     // [2, E]
  const float* msg = (const float*)d_in[2];
  const float* Wq  = (const float*)d_in[3];
  const float* bq  = (const float*)d_in[4];
  const float* Wk  = (const float*)d_in[5];
  const float* bk  = (const float*)d_in[6];
  const float* Wv  = (const float*)d_in[7];
  const float* bv  = (const float*)d_in[8];
  const float* We  = (const float*)d_in[9];
  const float* Ws  = (const float*)d_in[10];
  const float* bs  = (const float*)d_in[11];
  float* out = (float*)d_out;

  const int n = in_sizes[0] / DIN;
  const int E = in_sizes[1] / 2;

  // workspace layout: wt (f16, 32B-aligned first) | q | k | v | logit/p | m_enc | z
  _Float16* wt = (_Float16*)d_ws;                    // 5 * 128*128 halfs
  float* q     = (float*)(wt + (size_t)5 * DIN * HC);
  float* k     = q + (size_t)n * HC;
  float* v     = k + (size_t)n * HC;
  float* logit = v + (size_t)n * HC;
  unsigned* menc = (unsigned*)(logit + (size_t)E * 2);
  float* z     = (float*)(menc + (size_t)n * 2);

  const _Float16* wte = wt + (size_t)4 * DIN * HC;   // We is matrix index 4

  const int n2 = n * 2;
  k_cvt_w<<<(5 * DIN * HC + 255) / 256, 256, 0, stream>>>(Wq, Wk, Wv, Ws, We, wt);
  k_init<<<(n2 + 255) / 256, 256, 0, stream>>>(menc, z, n2);
  k_node_linear<<<(n + 63) / 64, 256, 0, stream>>>(
      x, wt, bq, bk, bv, bs, q, k, v, out, n);
  k_edge_logits<<<(E + 63) / 64, 256, 0, stream>>>(
      msg, wte, ei, q, k, logit, menc, E);
  k_softmax_z<<<(2 * E + 255) / 256, 256, 0, stream>>>(ei, menc, logit, z, E);
  k_edge_accum<<<(E + 63) / 64, 256, 0, stream>>>(
      msg, wte, ei, v, logit, z, out, E);
}
